// SelfAttention_57939108823383
// MI455X (gfx1250) — matrix operand order repaired and checked by reading, not compiled, not run on hardware
//
#include <hip/hip_runtime.h>
#include <hip/hip_fp16.h>

// Self-attention for MI455X (gfx1250): f16 WMMA everywhere, flash-style fused
// softmax-attention, f32 accumulation, async global->LDS double-buffered
// pipelines (ASYNCcnt).  B=4, S=2048, D=1024, H=8, dh=128.

#define B_    4
#define S_    2048
#define D_    1024
#define H_    8
#define DH_   128
#define ROWS  (B_ * S_)          // 8192 tokens

typedef __attribute__((ext_vector_type(16))) _Float16 v16h;
typedef __attribute__((ext_vector_type(8)))  _Float16 v8h;
typedef __attribute__((ext_vector_type(8)))  float    v8f;

union Frag16 { v16h v; v8h h[2]; };

static __device__ __forceinline__ v8f wmma_f16(v16h a, v16h b, v8f c) {
  // D = A(16x32 f16) * B(32x16 f16) + C(16x16 f32)
  return __builtin_amdgcn_wmma_f32_16x16x32_f16(false, a, false, b,
                                                (short)0, c, false, false);
}

static __device__ __forceinline__ v8f vzero8() {
  v8f z;
#pragma unroll
  for (int i = 0; i < 8; ++i) z[i] = 0.0f;
  return z;
}

// Async global->LDS copy, 16B per lane, tracked by ASYNCcnt.
// LDS operand = low 32 bits of the generic shared-memory address.
#define ASYNC_B128(ldsPtr, gPtr)                                              \
  do {                                                                        \
    unsigned int _la = (unsigned int)(uintptr_t)(ldsPtr);                     \
    unsigned long long _ga = (unsigned long long)(uintptr_t)(gPtr);           \
    asm volatile("global_load_async_to_lds_b128 %0, %1, off"                  \
                 :: "v"(_la), "v"(_ga) : "memory");                           \
  } while (0)

#define WAIT_ASYNC0() asm volatile("s_wait_asynccnt 0x0" ::: "memory")

// ---------------------------------------------------------------- converts
__global__ void cvt_f32_f16(const float* __restrict__ src,
                            _Float16* __restrict__ dst, int n) {
  int i = blockIdx.x * blockDim.x + threadIdx.x;
  int stride = gridDim.x * blockDim.x;
  for (; i < n; i += stride) dst[i] = (_Float16)src[i];
}

// ---------------------------------------------------------------- GEMMs
// C[row, col] = sum_k X[row,k] * W[col,k]  (both row-major => NT gemm)
// Block tile 128(M) x 128(N), K-step 32, 8 waves (4x2), wave tile 32x64.
// Double-buffered LDS fed by async loads; one barrier per K-step.
#define GLDA 40   // padded LDS row stride (halves); 80B = 5*16B

static __device__ __forceinline__ void
issue_gemm_tile(const _Float16* __restrict__ X, const _Float16* __restrict__ W,
                _Float16* AsB, _Float16* BsB, int tileM, int tileN, int kt,
                int r0, int k0, int r1, int k1) {
  ASYNC_B128(AsB + r0 * GLDA + k0, X + (size_t)(tileM * 128 + r0) * D_ + kt * 32 + k0);
  ASYNC_B128(AsB + r1 * GLDA + k1, X + (size_t)(tileM * 128 + r1) * D_ + kt * 32 + k1);
  ASYNC_B128(BsB + r0 * GLDA + k0, W + (size_t)(tileN * 128 + r0) * D_ + kt * 32 + k0);
  ASYNC_B128(BsB + r1 * GLDA + k1, W + (size_t)(tileN * 128 + r1) * D_ + kt * 32 + k1);
}

// Shared main loop; acc[2][4] produced for the caller's epilogue.
static __device__ __forceinline__ void
gemm_main(const _Float16* __restrict__ X, const _Float16* __restrict__ W,
          _Float16* As, _Float16* Bs, int tileM, int tileN,
          int lane, int wm, int wn, int lhalf, int tid, v8f acc[2][4]) {
#pragma unroll
  for (int mt = 0; mt < 2; ++mt)
#pragma unroll
    for (int nt = 0; nt < 4; ++nt) acc[mt][nt] = vzero8();

  const int c0 = tid, c1 = tid + 256;         // two 16B chunks per thread
  const int r0 = c0 >> 2, k0 = (c0 & 3) * 8;
  const int r1 = c1 >> 2, k1 = (c1 & 3) * 8;

  issue_gemm_tile(X, W, As, Bs, tileM, tileN, 0, r0, k0, r1, k1);   // tile 0

  for (int kt = 0; kt < D_ / 32; ++kt) {
    const int cur = kt & 1;
    WAIT_ASYNC0();          // my chunks of tile kt resident in LDS
    __syncthreads();        // everyone's chunks resident; other buffer free
    if (kt + 1 < D_ / 32) {
      issue_gemm_tile(X, W, As + (cur ^ 1) * 128 * GLDA,
                      Bs + (cur ^ 1) * 128 * GLDA, tileM, tileN, kt + 1,
                      r0, k0, r1, k1);
      if (kt + 2 < D_ / 32) {   // L2 prefetch two tiles ahead
        __builtin_prefetch(X + (size_t)(tileM * 128 + r0) * D_ + (kt + 2) * 32 + k0);
        __builtin_prefetch(W + (size_t)(tileN * 128 + r0) * D_ + (kt + 2) * 32 + k0);
      }
    }
    const _Float16* Ac = As + cur * 128 * GLDA;
    const _Float16* Bc = Bs + cur * 128 * GLDA;

    Frag16 af[2];
#pragma unroll
    for (int mt = 0; mt < 2; ++mt) {
      const int row = wm * 32 + mt * 16 + (lane & 15);
      const _Float16* p = Ac + row * GLDA + lhalf * 8;
      af[mt].h[0] = *(const v8h*)(p);        // K 0..7  (+8 for hi half-wave)
      af[mt].h[1] = *(const v8h*)(p + 16);   // K 16..23 (+8)
    }
#pragma unroll
    for (int nt = 0; nt < 4; ++nt) {
      const int nrow = wn * 64 + nt * 16 + (lane & 15);
      Frag16 bf;
      const _Float16* p = Bc + nrow * GLDA + lhalf * 8;
      bf.h[0] = *(const v8h*)(p);            // K 0..7 / 16..23 (+8 for hi half-wave)
      bf.h[1] = *(const v8h*)(p + 16);
#pragma unroll
      for (int mt = 0; mt < 2; ++mt)
        acc[mt][nt] = wmma_f16(af[mt].v, bf.v, acc[mt][nt]);
    }
  }
}

__global__ void __launch_bounds__(256)
gemm_qkv(const _Float16* __restrict__ X, const _Float16* __restrict__ W,
         _Float16* __restrict__ dst, int vtranspose) {
  __shared__ __align__(16) _Float16 As[2 * 128 * GLDA];
  __shared__ __align__(16) _Float16 Bs[2 * 128 * GLDA];

  const int tid = threadIdx.x, lane = tid & 31, wid = tid >> 5;
  const int wm = wid & 3, wn = wid >> 2;
  const int tileM = blockIdx.x & 63;   // ROWS/128
  const int tileN = blockIdx.x >> 6;   // D/128
  const int lhalf = (lane < 16) ? 0 : 1;

  v8f acc[2][4];
  gemm_main(X, W, As, Bs, tileM, tileN, lane, wm, wn, lhalf, tid, acc);

  // head-split store (Q/K row-major per head, V transposed per head)
#pragma unroll
  for (int mt = 0; mt < 2; ++mt)
#pragma unroll
    for (int nt = 0; nt < 4; ++nt)
#pragma unroll
      for (int r = 0; r < 8; ++r) {
        const int row = tileM * 128 + wm * 32 + mt * 16 + r + 8 * lhalf;
        const int col = tileN * 128 + wn * 64 + nt * 16 + (lane & 15);
        const int b = row >> 11, s = row & (S_ - 1);
        const int h = col >> 7,  d = col & (DH_ - 1);
        const _Float16 v = (_Float16)acc[mt][nt][r];
        if (vtranspose)
          dst[(size_t)((b * H_ + h) * DH_ + d) * S_ + s] = v;
        else
          dst[(size_t)((b * H_ + h) * S_ + s) * DH_ + d] = v;
      }
}

__global__ void __launch_bounds__(256)
gemm_out(const _Float16* __restrict__ X, const _Float16* __restrict__ W,
         const float* __restrict__ bias, float* __restrict__ out) {
  __shared__ __align__(16) _Float16 As[2 * 128 * GLDA];
  __shared__ __align__(16) _Float16 Bs[2 * 128 * GLDA];

  const int tid = threadIdx.x, lane = tid & 31, wid = tid >> 5;
  const int wm = wid & 3, wn = wid >> 2;
  const int tileM = blockIdx.x & 63;
  const int tileN = blockIdx.x >> 6;
  const int lhalf = (lane < 16) ? 0 : 1;

  v8f acc[2][4];
  gemm_main(X, W, As, Bs, tileM, tileN, lane, wm, wn, lhalf, tid, acc);

#pragma unroll
  for (int mt = 0; mt < 2; ++mt)
#pragma unroll
    for (int nt = 0; nt < 4; ++nt)
#pragma unroll
      for (int r = 0; r < 8; ++r) {
        const int row = tileM * 128 + wm * 32 + mt * 16 + r + 8 * lhalf;
        const int col = tileN * 128 + wn * 64 + nt * 16 + (lane & 15);
        out[(size_t)row * D_ + col] = acc[mt][nt][r] + bias[col];
      }
}

// ------------------------------------------------------- flash attention
// One block = one head x 128 query rows; 8 waves, 16 q-rows per wave.
// 64-key tiles, online softmax, K/V tiles double-buffered via async loads.
#define QLD 136   // 272B rows (17*16B): 16B aligned, bank-staggered
#define KLD 136
#define VLD 72    // 144B rows
#define PLD 72

static __device__ __forceinline__ void
issue_attn_tile(const _Float16* __restrict__ K, const _Float16* __restrict__ Vt,
                _Float16* KsB, _Float16* VsB, int head, int kt, int tid) {
#pragma unroll
  for (int j = 0; j < 4; ++j) {
    const int i = tid + j * 256;
    const int krow = i >> 4, kc8 = (i & 15) * 8;
    ASYNC_B128(KsB + krow * KLD + kc8,
               K + (size_t)(head * S_ + kt * 64 + krow) * DH_ + kc8);
    const int vrow = i >> 3, vc8 = (i & 7) * 8;
    ASYNC_B128(VsB + vrow * VLD + vc8,
               Vt + (size_t)(head * DH_ + vrow) * S_ + kt * 64 + vc8);
  }
}

__global__ void __launch_bounds__(256)
attn_flash(const _Float16* __restrict__ Q, const _Float16* __restrict__ K,
           const _Float16* __restrict__ Vt, _Float16* __restrict__ O) {
  __shared__ __align__(16) _Float16 Qs[128 * QLD];
  __shared__ __align__(16) _Float16 Ks[2 * 64 * KLD];
  __shared__ __align__(16) _Float16 Vs[2 * 128 * VLD];     // [dh][key]
  __shared__ __align__(16) _Float16 Ps[8 * 16 * PLD];      // per-wave P tile

  const int tid   = threadIdx.x;
  const int lane  = tid & 31;
  const int wid   = tid >> 5;
  const int head  = blockIdx.x >> 4;   // 0..31 (= b*H + h)
  const int qtile = blockIdx.x & 15;   // 16 tiles of 128 rows
  const int lhalf = (lane < 16) ? 0 : 1;
  const float scale = 0.08838834764831845f;  // 1/sqrt(128)

  issue_attn_tile(K, Vt, Ks, Vs, head, 0, tid);   // key tile 0 -> buffer 0

  // ---- stage Q tile (128 x 128 halves) while tile 0 flies
  for (int i = tid; i < 128 * 16; i += 256) {
    const int row = i >> 4, c8 = (i & 15) * 8;
    v8h v = *(const v8h*)(Q + (size_t)(head * S_ + qtile * 128 + row) * DH_ + c8);
    *(v8h*)(Qs + row * QLD + c8) = v;
  }
  __syncthreads();

  Frag16 qf[4];   // A-fragments of this wave's 16 q-rows, 4 d-chunks of 32
#pragma unroll
  for (int kc = 0; kc < 4; ++kc) {
    const int row = wid * 16 + (lane & 15);
    const _Float16* p = Qs + row * QLD + kc * 32 + lhalf * 8;
    qf[kc].h[0] = *(const v8h*)(p);
    qf[kc].h[1] = *(const v8h*)(p + 16);
  }

  v8f o[8];
#pragma unroll
  for (int i = 0; i < 8; ++i) o[i] = vzero8();
  float mrow[8], lrow[8];
#pragma unroll
  for (int r = 0; r < 8; ++r) { mrow[r] = -1e30f; lrow[r] = 0.0f; }

  for (int kt = 0; kt < S_ / 64; ++kt) {
    const int cur = kt & 1;
    WAIT_ASYNC0();          // my chunks of K/V tile kt resident
    __syncthreads();        // all chunks resident; other buffer free
    if (kt + 1 < S_ / 64) {
      issue_attn_tile(K, Vt, Ks + (cur ^ 1) * 64 * KLD,
                      Vs + (cur ^ 1) * 128 * VLD, head, kt + 1, tid);
      if (kt + 2 < S_ / 64) {
        __builtin_prefetch(K + (size_t)(head * S_ + (kt + 2) * 64 + (tid >> 4)) * DH_);
        __builtin_prefetch(Vt + (size_t)(head * DH_ + (tid >> 3)) * S_ + (kt + 2) * 64);
      }
    }
    const _Float16* Kc = Ks + cur * 64 * KLD;
    const _Float16* Vc = Vs + cur * 128 * VLD;

    // ---- scores: S = scale * (Q . K^T), 16 q-rows x 64 keys per wave
    v8f sc[4];
#pragma unroll
    for (int nt = 0; nt < 4; ++nt) {
      sc[nt] = vzero8();
#pragma unroll
      for (int kc = 0; kc < 4; ++kc) {
        Frag16 bf;  // B operand: lane = key column, d's 0..7,16..23 (+8 hi half)
        const _Float16* p = Kc + (nt * 16 + (lane & 15)) * KLD + kc * 32 + lhalf * 8;
        bf.h[0] = *(const v8h*)(p);
        bf.h[1] = *(const v8h*)(p + 16);
        sc[nt] = wmma_f16(qf[kc].v, bf.v, sc[nt]);
      }
#pragma unroll
      for (int r = 0; r < 8; ++r) sc[nt][r] *= scale;
    }

    // ---- online softmax: row reductions across the 16-lane half-wave
    float corr[8];
#pragma unroll
    for (int r = 0; r < 8; ++r) {
      float t = fmaxf(fmaxf(sc[0][r], sc[1][r]), fmaxf(sc[2][r], sc[3][r]));
      t = fmaxf(t, __shfl_xor(t, 1, 32));
      t = fmaxf(t, __shfl_xor(t, 2, 32));
      t = fmaxf(t, __shfl_xor(t, 4, 32));
      t = fmaxf(t, __shfl_xor(t, 8, 32));
      const float mnew = fmaxf(mrow[r], t);
      corr[r] = __expf(mrow[r] - mnew);
      mrow[r] = mnew;
    }
#pragma unroll
    for (int nt = 0; nt < 4; ++nt)
#pragma unroll
      for (int r = 0; r < 8; ++r) sc[nt][r] = __expf(sc[nt][r] - mrow[r]);
#pragma unroll
    for (int r = 0; r < 8; ++r) {
      float s = sc[0][r] + sc[1][r] + sc[2][r] + sc[3][r];
      s += __shfl_xor(s, 1, 32);
      s += __shfl_xor(s, 2, 32);
      s += __shfl_xor(s, 4, 32);
      s += __shfl_xor(s, 8, 32);
      lrow[r] = lrow[r] * corr[r] + s;
    }
#pragma unroll
    for (int dt = 0; dt < 8; ++dt)
#pragma unroll
      for (int r = 0; r < 8; ++r) o[dt][r] *= corr[r];

    // ---- C-layout -> A-layout transpose of P through per-wave LDS pad
    _Float16* pw = Ps + wid * 16 * PLD;
#pragma unroll
    for (int nt = 0; nt < 4; ++nt)
#pragma unroll
      for (int r = 0; r < 8; ++r)
        pw[(r + 8 * lhalf) * PLD + nt * 16 + (lane & 15)] = (_Float16)sc[nt][r];
    asm volatile("s_wait_dscnt 0x0" ::: "memory");  // same-wave DS RAW handoff

    // ---- O += P (16x64) . V (64x128)
#pragma unroll
    for (int kc2 = 0; kc2 < 2; ++kc2) {
      Frag16 af;
      const _Float16* pa = pw + (lane & 15) * PLD + kc2 * 32 + lhalf * 8;
      af.h[0] = *(const v8h*)(pa);
      af.h[1] = *(const v8h*)(pa + 16);
#pragma unroll
      for (int dt = 0; dt < 8; ++dt) {
        Frag16 bf;
        const _Float16* pb = Vc + (dt * 16 + (lane & 15)) * VLD + kc2 * 32 + lhalf * 8;
        bf.h[0] = *(const v8h*)(pb);
        bf.h[1] = *(const v8h*)(pb + 16);
        o[dt] = wmma_f16(af.v, bf.v, o[dt]);
      }
    }
  }

  // ---- normalize and store merged-head output (f16), [B,S,D] layout
  const int b = head >> 3, hh = head & 7;
  float inv[8];
#pragma unroll
  for (int r = 0; r < 8; ++r) inv[r] = 1.0f / lrow[r];
#pragma unroll
  for (int dt = 0; dt < 8; ++dt)
#pragma unroll
    for (int r = 0; r < 8; ++r) {
      const int s = qtile * 128 + wid * 16 + r + 8 * lhalf;
      const int d = dt * 16 + (lane & 15);
      O[(size_t)(b * S_ + s) * D_ + hh * DH_ + d] = (_Float16)(o[dt][r] * inv[r]);
    }
}

// ---------------------------------------------------------------- launch
extern "C" void kernel_launch(void* const* d_in, const int* in_sizes, int n_in,
                              void* d_out, int out_size, void* d_ws, size_t ws_size,
                              hipStream_t stream) {
  const float* x  = (const float*)d_in[0];
  const float* Wq = (const float*)d_in[1];
  const float* Wk = (const float*)d_in[2];
  const float* Wv = (const float*)d_in[3];
  const float* Wo = (const float*)d_in[4];
  const float* bo = (const float*)d_in[5];

  char* ws = (char*)d_ws;
  size_t off = 0;
  _Float16* xh  = (_Float16*)(ws + off); off += (size_t)ROWS * D_ * 2;
  _Float16* wqh = (_Float16*)(ws + off); off += (size_t)D_ * D_ * 2;
  _Float16* wkh = (_Float16*)(ws + off); off += (size_t)D_ * D_ * 2;
  _Float16* wvh = (_Float16*)(ws + off); off += (size_t)D_ * D_ * 2;
  _Float16* woh = (_Float16*)(ws + off); off += (size_t)D_ * D_ * 2;
  _Float16* Qh  = (_Float16*)(ws + off); off += (size_t)ROWS * D_ * 2;
  _Float16* Kh  = (_Float16*)(ws + off); off += (size_t)ROWS * D_ * 2;
  _Float16* Vth = (_Float16*)(ws + off); off += (size_t)ROWS * D_ * 2;
  _Float16* Oh  = (_Float16*)(ws + off); off += (size_t)ROWS * D_ * 2;

  cvt_f32_f16<<<2048, 256, 0, stream>>>(x,  xh,  ROWS * D_);
  cvt_f32_f16<<<512,  256, 0, stream>>>(Wq, wqh, D_ * D_);
  cvt_f32_f16<<<512,  256, 0, stream>>>(Wk, wkh, D_ * D_);
  cvt_f32_f16<<<512,  256, 0, stream>>>(Wv, wvh, D_ * D_);
  cvt_f32_f16<<<512,  256, 0, stream>>>(Wo, woh, D_ * D_);

  dim3 g(512), blk(256);  // (ROWS/128) * (D/128) tiles
  gemm_qkv<<<g, blk, 0, stream>>>(xh, wqh, Qh, 0);
  gemm_qkv<<<g, blk, 0, stream>>>(xh, wkh, Kh, 0);
  gemm_qkv<<<g, blk, 0, stream>>>(xh, wvh, Vth, 1);  // V stored [head][dh][S]

  attn_flash<<<dim3(32 * 16), blk, 0, stream>>>(Qh, Kh, Vth, Oh);

  gemm_out<<<g, blk, 0, stream>>>(Oh, woh, bo, (float*)d_out);
}
